// MiniMaxText01SparseMoeBlock_13907104105123
// MI455X (gfx1250) — compile-verified, matrix operand order checked by reading, not run on hardware
//
#include <hip/hip_runtime.h>
#include <hip/hip_bf16.h>
#include <math.h>

#define NUM_EXPERTS 8
#define HIDDEN 2048
#define INTER 5120
#define TOKENS 4096  // B*S = 2*2048
#define IC 2560      // intermediate-dimension chunk held in workspace (2 chunks)
#define KSTEP 32     // WMMA K per step (bf16 16x16x32)
#define BM 128       // block tile: tokens
#define BN 64        // block tile: columns

typedef __bf16 bf16_t;
typedef __attribute__((ext_vector_type(16))) __bf16 v16bf;
typedef __attribute__((ext_vector_type(8)))  __bf16 v8bf;
typedef __attribute__((ext_vector_type(2)))  __bf16 v2bf;
typedef __attribute__((ext_vector_type(8)))  float  v8f;
typedef __attribute__((ext_vector_type(4)))  float  v4f;

__device__ __forceinline__ v8f wmma_bf16(v16bf a, v16bf b, v8f c) {
    // D = A(16x32 bf16) * B(32x16 bf16) + C(16x16 f32)
    return __builtin_amdgcn_wmma_f32_16x16x32_bf16(
        /*neg_a=*/false, a, /*neg_b=*/false, b,
        /*c_mod=*/(short)0, c, /*reuse_a=*/false, /*reuse_b=*/false);
}

// Non-temporal streaming load of one float4 (weights are read exactly once).
__device__ __forceinline__ v4f nt_load4(const float* __restrict__ p) {
    return __builtin_nontemporal_load((const v4f*)p);
}

// A-fragment from fp32 row-major matrix; p = row pointer already offset to kb.
// Lane m = lane&15 (handled by caller's row pointer); lane>=16 takes K+8 halves.
__device__ __forceinline__ v16bf frag_from_f32(const float* __restrict__ p, int lane) {
    const int klo = (lane >> 4) << 3;  // 0 or 8
    const v4f* q0 = (const v4f*)(p + klo);        // K = klo..klo+7
    const v4f* q1 = (const v4f*)(p + 16 + klo);   // K = 16+klo..16+klo+7
    v4f f0 = q0[0], f1 = q0[1], f2 = q1[0], f3 = q1[1];
    v16bf r;
#pragma unroll
    for (int j = 0; j < 4; ++j) {
        r[j]      = (__bf16)f0[j];
        r[4 + j]  = (__bf16)f1[j];
        r[8 + j]  = (__bf16)f2[j];
        r[12 + j] = (__bf16)f3[j];
    }
    return r;
}

// A-fragment from bf16 row-major matrix; p = row pointer already offset to kb.
__device__ __forceinline__ v16bf frag_from_bf16(const bf16_t* __restrict__ p, int lane) {
    const int klo = (lane >> 4) << 3;
    v8bf lo = *(const v8bf*)(p + klo);
    v8bf hi = *(const v8bf*)(p + 16 + klo);
    v16bf r;
#pragma unroll
    for (int j = 0; j < 8; ++j) { r[j] = lo[j]; r[8 + j] = hi[j]; }
    return r;
}

// B-fragment from LDS tile stored transposed as lds[n][KSTEP] (n in 0..BN-1).
// Lane n = nOff + (lane&15); contiguous K runs -> ds_load_b128 pairs.
__device__ __forceinline__ v16bf frag_b_lds(const bf16_t* lds, int nOff, int lane) {
    const int n = nOff + (lane & 15);
    const int klo = (lane >> 4) << 3;
    const bf16_t* p = lds + n * KSTEP + klo;
    v8bf lo = *(const v8bf*)p;         // K = klo..klo+7
    v8bf hi = *(const v8bf*)(p + 16);  // K = 16+klo..
    v16bf r;
#pragma unroll
    for (int j = 0; j < 8; ++j) { r[j] = lo[j]; r[8 + j] = hi[j]; }
    return r;
}

// Convert two staged fp32 rows (k, k+1) at columns n..n+3 and store packed
// bf16 pairs into the transposed LDS tile: one cvt_pk + one ds_store_b32 each.
__device__ __forceinline__ void stage_to_lds(bf16_t* lds, int k, int n, v4f ra, v4f rb) {
#pragma unroll
    for (int j = 0; j < 4; ++j) {
        v2bf pr;
        pr[0] = (__bf16)ra[j];   // (k,   n+j)
        pr[1] = (__bf16)rb[j];   // (k+1, n+j)
        *(v2bf*)(lds + (n + j) * KSTEP + k) = pr;  // 4B aligned (k even)
    }
}

// ---------------- Router: logits, top-2, softmax -> coef[T][8] ----------------
__global__ __launch_bounds__(256) void moe_router_kernel(
    const float* __restrict__ x, const float* __restrict__ gw,
    float* __restrict__ logits, float* __restrict__ coef) {
    const int lane  = threadIdx.x & 31;
    const int token = blockIdx.x * 8 + (threadIdx.x >> 5);
    const float* xp = x + (size_t)token * HIDDEN;

    float s[NUM_EXPERTS];
#pragma unroll
    for (int e = 0; e < NUM_EXPERTS; ++e) s[e] = 0.0f;

    for (int h = lane; h < HIDDEN; h += 32) {
        float xv = xp[h];
        const float* g = gw + (size_t)h * NUM_EXPERTS;
#pragma unroll
        for (int e = 0; e < NUM_EXPERTS; ++e) s[e] = fmaf(xv, g[e], s[e]);
    }
#pragma unroll
    for (int off = 16; off > 0; off >>= 1) {
#pragma unroll
        for (int e = 0; e < NUM_EXPERTS; ++e) s[e] += __shfl_xor(s[e], off, 32);
    }
    if (lane == 0) {
#pragma unroll
        for (int e = 0; e < NUM_EXPERTS; ++e) logits[(size_t)token * NUM_EXPERTS + e] = s[e];
        int b1 = 0; float v1 = s[0];
#pragma unroll
        for (int e = 1; e < NUM_EXPERTS; ++e) if (s[e] > v1) { v1 = s[e]; b1 = e; }
        int b2 = -1; float v2 = -INFINITY;
#pragma unroll
        for (int e = 0; e < NUM_EXPERTS; ++e)
            if (e != b1 && s[e] > v2) { v2 = s[e]; b2 = e; }
        float p1 = 1.0f / (1.0f + __expf(v2 - v1));
        float p2 = 1.0f - p1;
        float* cp = coef + (size_t)token * NUM_EXPERTS;
#pragma unroll
        for (int e = 0; e < NUM_EXPERTS; ++e) cp[e] = 0.0f;
        cp[b1] = p1;
        cp[b2] = p2;
    }
}

// ---- Gated GEMM: gbuf[t][ic] = coef[t][e] * silu(x@w1) * (x@w3), bf16 out ----
// Double-buffered LDS weight tiles; NT weight streaming; prefetch 2 tiles ahead.
__global__ __launch_bounds__(256) void moe_gated_kernel(
    const float* __restrict__ x, const float* __restrict__ w1e,
    const float* __restrict__ w3e, const float* __restrict__ coef,
    bf16_t* __restrict__ gbuf, int i0, int expert) {
    __shared__ bf16_t ldsB1[2][BN * KSTEP];
    __shared__ bf16_t ldsB3[2][BN * KSTEP];
    const int tid   = threadIdx.x;
    const int lane  = tid & 31;
    const int wave  = tid >> 5;
    const int waveM = wave & 3;   // 4 waves along M (32 tokens each)
    const int waveN = wave >> 2;  // 2 waves along N (32 cols each)
    const int rowBase = blockIdx.x * BM + waveM * 32;
    const int colBase = blockIdx.y * BN;  // chunk-local column base

    v8f acc1[2][2] = {{{0}}};
    v8f acc3[2][2] = {{{0}}};

    const int m = lane & 15;
    const float* a0p = x + (size_t)(rowBase + m) * HIDDEN;
    const float* a1p = x + (size_t)(rowBase + 16 + m) * HIDDEN;

    // Fill assignment: thread -> rows (fk, fk+1), columns fn..fn+3 of the tile.
    const int fk = (tid >> 4) << 1;   // 0,2,...,30
    const int fn = (tid & 15) << 2;   // 0,4,...,60
    const size_t wstep = (size_t)KSTEP * INTER;
    const float* w1p = w1e + (size_t)fk * INTER + (i0 + colBase + fn);
    const float* w3p = w3e + (size_t)fk * INTER + (i0 + colBase + fn);

    // Prologue: stage tile 0 into buffer 0.
    {
        v4f s1a = nt_load4(w1p), s1b = nt_load4(w1p + INTER);
        v4f s3a = nt_load4(w3p), s3b = nt_load4(w3p + INTER);
        stage_to_lds(ldsB1[0], fk, fn, s1a, s1b);
        stage_to_lds(ldsB3[0], fk, fn, s3a, s3b);
    }
    __syncthreads();

    const int NIT = HIDDEN / KSTEP;
    for (int it = 0; it < NIT; ++it) {
        const int cur = it & 1;
        const bool hasNext = (it + 1 < NIT);
        v4f s1a, s1b, s3a, s3b;
        if (hasNext) {  // issue next-tile streaming loads before the math
            w1p += wstep; w3p += wstep;
            s1a = nt_load4(w1p); s1b = nt_load4(w1p + INTER);
            s3a = nt_load4(w3p); s3b = nt_load4(w3p + INTER);
            __builtin_prefetch(w1p + wstep, 0, 0);  // 2 tiles ahead
            __builtin_prefetch(w3p + wstep, 0, 0);
        }

        const int kb = it * KSTEP;
        v16bf a[2];
        a[0] = frag_from_f32(a0p + kb, lane);
        a[1] = frag_from_f32(a1p + kb, lane);
#pragma unroll
        for (int sn = 0; sn < 2; ++sn) {
            int nOff = waveN * 32 + sn * 16;
            v16bf b1 = frag_b_lds(ldsB1[cur], nOff, lane);
            v16bf b3 = frag_b_lds(ldsB3[cur], nOff, lane);
#pragma unroll
            for (int sm = 0; sm < 2; ++sm) {
                acc1[sm][sn] = wmma_bf16(a[sm], b1, acc1[sm][sn]);
                acc3[sm][sn] = wmma_bf16(a[sm], b3, acc3[sm][sn]);
            }
        }

        if (hasNext) {  // convert + store next tile into the other buffer
            stage_to_lds(ldsB1[cur ^ 1], fk, fn, s1a, s1b);
            stage_to_lds(ldsB3[cur ^ 1], fk, fn, s3a, s3b);
        }
        __syncthreads();
    }

    // Epilogue: coef * silu(a1) * a3 -> bf16 gated chunk
    const int halfOff = (lane >> 4) << 3;  // C/D: M = r + 8*(lane>=16)
#pragma unroll
    for (int sm = 0; sm < 2; ++sm) {
#pragma unroll
        for (int r = 0; r < 8; ++r) {
            int t = rowBase + sm * 16 + r + halfOff;
            float c = coef[(size_t)t * NUM_EXPERTS + expert];
#pragma unroll
            for (int sn = 0; sn < 2; ++sn) {
                float g  = acc1[sm][sn][r];
                float sg = g / (1.0f + __expf(-g));  // silu
                float v  = c * sg * acc3[sm][sn][r];
                int n = waveN * 32 + sn * 16 + (lane & 15);
                gbuf[(size_t)t * IC + colBase + n] = (__bf16)v;
            }
        }
    }
}

// ------- Down GEMM: out[t][h] += gbuf[t][:] @ w2[i0+:, h]  (f32 accum) -------
__global__ __launch_bounds__(256) void moe_down_kernel(
    const bf16_t* __restrict__ gbuf, const float* __restrict__ w2e,
    float* __restrict__ out, int i0) {
    __shared__ bf16_t ldsB[2][BN * KSTEP];
    const int tid   = threadIdx.x;
    const int lane  = tid & 31;
    const int wave  = tid >> 5;
    const int waveM = wave & 3;
    const int waveN = wave >> 2;
    const int rowBase = blockIdx.x * BM + waveM * 32;
    const int colBase = blockIdx.y * BN;  // h base

    v8f acc[2][2] = {{{0}}};

    const int m = lane & 15;
    const bf16_t* a0p = gbuf + (size_t)(rowBase + m) * IC;
    const bf16_t* a1p = gbuf + (size_t)(rowBase + 16 + m) * IC;

    const int fk = (tid >> 4) << 1;
    const int fn = (tid & 15) << 2;
    const size_t wstep = (size_t)KSTEP * HIDDEN;
    const float* w2p = w2e + (size_t)(i0 + fk) * HIDDEN + (colBase + fn);

    {
        v4f sa = nt_load4(w2p), sb = nt_load4(w2p + HIDDEN);
        stage_to_lds(ldsB[0], fk, fn, sa, sb);
    }
    __syncthreads();

    const int NIT = IC / KSTEP;
    for (int it = 0; it < NIT; ++it) {
        const int cur = it & 1;
        const bool hasNext = (it + 1 < NIT);
        v4f sa, sb;
        if (hasNext) {
            w2p += wstep;
            sa = nt_load4(w2p); sb = nt_load4(w2p + HIDDEN);
            __builtin_prefetch(w2p + wstep, 0, 0);
        }

        const int kb = it * KSTEP;
        v16bf a[2];
        a[0] = frag_from_bf16(a0p + kb, lane);
        a[1] = frag_from_bf16(a1p + kb, lane);
#pragma unroll
        for (int sn = 0; sn < 2; ++sn) {
            int nOff = waveN * 32 + sn * 16;
            v16bf b = frag_b_lds(ldsB[cur], nOff, lane);
#pragma unroll
            for (int sm = 0; sm < 2; ++sm)
                acc[sm][sn] = wmma_bf16(a[sm], b, acc[sm][sn]);
        }

        if (hasNext) stage_to_lds(ldsB[cur ^ 1], fk, fn, sa, sb);
        __syncthreads();
    }

    // Accumulate into out (unique tile ownership per launch; stream-ordered
    // across the 16 (expert, chunk) launches; out stays L2-resident).
    const int halfOff = (lane >> 4) << 3;
#pragma unroll
    for (int sm = 0; sm < 2; ++sm) {
#pragma unroll
        for (int r = 0; r < 8; ++r) {
            int t = rowBase + sm * 16 + r + halfOff;
#pragma unroll
            for (int sn = 0; sn < 2; ++sn) {
                int h = colBase + waveN * 32 + sn * 16 + (lane & 15);
                float* o = out + (size_t)t * HIDDEN + h;
                *o = *o + acc[sm][sn][r];
            }
        }
    }
}

extern "C" void kernel_launch(void* const* d_in, const int* in_sizes, int n_in,
                              void* d_out, int out_size, void* d_ws, size_t ws_size,
                              hipStream_t stream) {
    (void)in_sizes; (void)n_in; (void)out_size; (void)ws_size;
    const float* x      = (const float*)d_in[0];  // (B,S,H) fp32
    const float* gate_w = (const float*)d_in[1];  // (H,E)
    const float* w1     = (const float*)d_in[2];  // (E,H,I)
    const float* w3     = (const float*)d_in[3];  // (E,H,I)
    const float* w2     = (const float*)d_in[4];  // (E,I,H)

    float* out    = (float*)d_out;                        // (B,S,H)
    float* logits = out + (size_t)TOKENS * HIDDEN;        // (B,S,E)

    float*  coef = (float*)d_ws;                                          // T*8 f32
    bf16_t* gbuf = (bf16_t*)((char*)d_ws + (size_t)TOKENS * NUM_EXPERTS * sizeof(float)); // T*IC bf16

    hipMemsetAsync(out, 0, (size_t)TOKENS * HIDDEN * sizeof(float), stream);

    moe_router_kernel<<<TOKENS / 8, 256, 0, stream>>>(x, gate_w, logits, coef);

    for (int e = 0; e < NUM_EXPERTS; ++e) {
        const float* w1e = w1 + (size_t)e * HIDDEN * INTER;
        const float* w3e = w3 + (size_t)e * HIDDEN * INTER;
        const float* w2e = w2 + (size_t)e * INTER * HIDDEN;
        for (int c = 0; c < INTER / IC; ++c) {
            int i0 = c * IC;
            moe_gated_kernel<<<dim3(TOKENS / BM, IC / BN), 256, 0, stream>>>(
                x, w1e, w3e, coef, gbuf, i0, e);
            moe_down_kernel<<<dim3(TOKENS / BM, HIDDEN / BN), 256, 0, stream>>>(
                gbuf, w2e, out, i0);
        }
    }
}
